// NoisyTopKGating_55379308314952
// MI455X (gfx1250) — matrix operand order repaired and checked by reading, not compiled, not run on hardware
//
#include <hip/hip_runtime.h>
#include <hip/hip_bf16.h>
#include <math.h>

#define TOKENS     16384
#define HIDDEN     2048
#define NEXP       64
#define NCOL       128          // gate experts (64) ++ noise experts (64)
#define MTILE      128          // tokens per block
#define LDS_STRIDE 132          // 128 + 4 pad: avoids 64-bank conflicts on row reads
#define LDS_BYTES  (MTILE * LDS_STRIDE * 4)

typedef __bf16 v16bf __attribute__((ext_vector_type(16)));
typedef float  v8f   __attribute__((ext_vector_type(8)));
typedef float  v4f   __attribute__((ext_vector_type(4)));

// ---------------------------------------------------------------------------
// Kernel 1: convert/concat weights to bf16 [NCOL][HIDDEN] and zero accumulators
// accum layout: [0..63]=importance, [64..127]=loads, [128]=sum(lse^2)
// ---------------------------------------------------------------------------
__global__ void __launch_bounds__(256)
prep_weights(const float* __restrict__ wg, const float* __restrict__ wn,
             unsigned short* __restrict__ wb_u16, float* __restrict__ accum) {
    __bf16* wb = (__bf16*)wb_u16;
    int idx = blockIdx.x * 256 + threadIdx.x;      // [0, NCOL*HIDDEN)
    int row = idx >> 11;                            // /HIDDEN
    int col = idx & (HIDDEN - 1);
    float v = (row < NEXP) ? wg[row * HIDDEN + col] : wn[(row - NEXP) * HIDDEN + col];
    wb[idx] = (__bf16)v;
    if (blockIdx.x == 0 && threadIdx.x < 129) accum[threadIdx.x] = 0.0f;
}

// ---------------------------------------------------------------------------
// A-fragment loader: 16-bit A 16x32, contiguous k order used here (same as B):
//   lanes 0-15 : K 0..15 contiguous                       (row M = lane)
//   lanes16-31 : K 16..31 contiguous                      (row M = lane-16)
// Load 2x32B of fp32 x, convert to bf16 (packed cvt).
// ---------------------------------------------------------------------------
__device__ __forceinline__ v16bf load_a16(const float* __restrict__ xr, int k, int lHi) {
    const v4f* p0 = (const v4f*)(xr + k + lHi * 16);
    const v4f* p1 = (const v4f*)(xr + k + lHi * 16 + 8);
    v4f f0 = p0[0], f1 = p0[1], f2 = p1[0], f3 = p1[1];
    v16bf a;
#pragma unroll
    for (int i = 0; i < 4; ++i) {
        a[i]      = (__bf16)f0[i];
        a[4 + i]  = (__bf16)f1[i];
        a[8 + i]  = (__bf16)f2[i];
        a[12 + i] = (__bf16)f3[i];
    }
    return a;
}

// ---------------------------------------------------------------------------
// Kernel 2: fused dual-GEMM (clean|noise logits) + noisy top-2 gating epilogue
// Block = 256 threads = 8 wave32. Tile: 128 tokens x 128 cols, K step = 32.
// Wave (wr = w>>1 in 0..3, wc = w&1): rows [wr*32, wr*32+32), cols [wc*64, wc*64+64)
//   -> 2 row-tiles x 4 col-tiles = 8 x v_wmma_f32_16x16x32_bf16 per K step.
// ---------------------------------------------------------------------------
__global__ void __launch_bounds__(256)
moe_router(const float* __restrict__ x, const unsigned short* __restrict__ wb_u16,
           const float* __restrict__ noise, float* __restrict__ gates,
           float* __restrict__ accum) {
    extern __shared__ float L[];                   // [MTILE][LDS_STRIDE] logits
    const __bf16* wb = (const __bf16*)wb_u16;

    const int lane = threadIdx.x & 31;
    const int wv   = threadIdx.x >> 5;
    const int wr   = wv >> 1;                      // 0..3
    const int wc   = wv & 1;                       // 0..1
    const int l15  = lane & 15;
    const int lHi  = lane >> 4;

    const int rowBase = blockIdx.x * MTILE;
    const float* xr0 = x + (size_t)(rowBase + wr * 32 + l15) * HIDDEN;
    const float* xr1 = xr0 + (size_t)16 * HIDDEN;

    // B-fragment base pointers: 16-bit B 32x16, contiguous k order as in A — lane n holds column n,
    // lanes 0-15 K=0..15, lanes 16-31 K=16..31 (contiguous 32B per lane).
    const __bf16* wp[4];
#pragma unroll
    for (int j = 0; j < 4; ++j) {
        int ecol = (wc * 4 + j) * 16 + l15;
        wp[j] = wb + (size_t)ecol * HIDDEN + lHi * 16;
    }

    v8f acc[2][4] = {};

    for (int k = 0; k < HIDDEN; k += 32) {
        v16bf a0 = load_a16(xr0, k, lHi);
        v16bf a1 = load_a16(xr1, k, lHi);
#pragma unroll
        for (int j = 0; j < 4; ++j) {
            v16bf b = *(const v16bf*)(wp[j] + k);
            acc[0][j] = __builtin_amdgcn_wmma_f32_16x16x32_bf16(
                false, a0, false, b, (short)0, acc[0][j], false, false);
            acc[1][j] = __builtin_amdgcn_wmma_f32_16x16x32_bf16(
                false, a1, false, b, (short)0, acc[1][j], false, false);
        }
        // stream next x chunks toward the WGP (global_prefetch_b8)
        __builtin_prefetch(xr0 + k + 512, 0, 1);
        __builtin_prefetch(xr1 + k + 512, 0, 1);
    }

    // C/D f32 16x16 layout: VGPR v, lanes 0-15 -> M=v, lanes 16-31 -> M=v+8.
#pragma unroll
    for (int i = 0; i < 2; ++i) {
#pragma unroll
        for (int j = 0; j < 4; ++j) {
            int N = (wc * 4 + j) * 16 + l15;
#pragma unroll
            for (int v = 0; v < 8; ++v) {
                int M = wr * 32 + i * 16 + lHi * 8 + v;
                L[M * LDS_STRIDE + N] = acc[i][j][v];
            }
        }
    }
    __syncthreads();

    // -------- per-token gating epilogue: one thread per token --------------
    if (threadIdx.x < MTILE) {
        const int r  = threadIdx.x;
        const int rg = rowBase + r;
        const float* Lr = L + r * LDS_STRIDE;                 // [0..63]=clean, [64..127]=noise logit
        const float* nz = noise + (size_t)rg * NEXP;

        float m1 = -INFINITY, m2 = -INFINITY, cmax = -INFINITY;
        int   i1 = 0, i2 = 0;
        for (int e = 0; e < NEXP; ++e) {
            float c  = Lr[e];
            float nl = Lr[NEXP + e];
            float sd = (nl > 20.0f) ? nl : log1pf(__expf(nl)); // softplus
            sd += 1e-10f;
            float l = fmaf(nz[e], sd, c);                      // noisy logit
            if (l > m1)      { m2 = m1; i2 = i1; m1 = l; i1 = e; }
            else if (l > m2) { m2 = l;  i2 = e; }
            cmax = fmaxf(cmax, c);
        }
        float s = 0.0f;
        for (int e = 0; e < NEXP; ++e) s += __expf(Lr[e] - cmax);
        float lse = cmax + __logf(s);                          // lse of clean logits

        float g1 = 1.0f / (1.0f + __expf(m2 - m1));            // softmax over top-2
        float g2 = 1.0f - g1;

        float* orow = gates + (size_t)rg * NEXP;
#pragma unroll
        for (int q = 0; q < 16; ++q) {
            v4f v;
#pragma unroll
            for (int c = 0; c < 4; ++c) {
                int e = q * 4 + c;
                v[c] = (e == i1) ? g1 : ((e == i2) ? g2 : 0.0f);
            }
            *(v4f*)(orow + q * 4) = v;                         // global_store_b128
        }
        atomicAdd(&accum[i1], g1);                             // importance
        atomicAdd(&accum[i2], g2);
        if (g1 > 0.0f) atomicAdd(&accum[64 + i1], 1.0f);       // loads
        if (g2 > 0.0f) atomicAdd(&accum[64 + i2], 1.0f);
        atomicAdd(&accum[128], lse * lse);                     // z-loss accumulator
    }
}

// ---------------------------------------------------------------------------
// Kernel 3: combine global reductions into the scalar loss
// ---------------------------------------------------------------------------
__global__ void finalize_loss(const float* __restrict__ accum, float* __restrict__ out_loss) {
    if (threadIdx.x == 0 && blockIdx.x == 0) {
        float s = 0.0f;
        for (int e = 0; e < NEXP; ++e) s += accum[e] * accum[64 + e];
        float nt = (float)TOKENS;
        float loss = 0.01f * ((float)NEXP * s / (nt * nt)) + 0.001f * (accum[128] / nt);
        *out_loss = loss;
    }
}

// ---------------------------------------------------------------------------
extern "C" void kernel_launch(void* const* d_in, const int* in_sizes, int n_in,
                              void* d_out, int out_size, void* d_ws, size_t ws_size,
                              hipStream_t stream) {
    (void)in_sizes; (void)n_in; (void)out_size; (void)ws_size;
    const float* x  = (const float*)d_in[0];
    const float* wg = (const float*)d_in[1];
    const float* wn = (const float*)d_in[2];
    const float* nz = (const float*)d_in[3];
    float* out = (float*)d_out;

    unsigned short* wb = (unsigned short*)d_ws;                          // bf16 [128][2048]
    float* accum = (float*)((char*)d_ws + (size_t)NCOL * HIDDEN * sizeof(unsigned short));

    prep_weights<<<(NCOL * HIDDEN) / 256, 256, 0, stream>>>(wg, wn, wb, accum);
    moe_router<<<TOKENS / MTILE, 256, LDS_BYTES, stream>>>(x, wb, nz, out, accum);
    finalize_loss<<<1, 64, 0, stream>>>(accum, out + (size_t)TOKENS * NEXP);
}
